// WindowAttention_S_28312424415802
// MI455X (gfx1250) — compile-verified, matrix operand order checked by reading, not run on hardware
//
#include <hip/hip_runtime.h>

// Fused Swin-V2 window attention for gfx1250 (wave32, WMMA f16 16x16x32).
// One workgroup (256 threads = 8 waves) per window; everything per-window in LDS.

#define TOK  64
#define DIMC 192
#define NH   6
#define HD   32

typedef _Float16 v16h __attribute__((ext_vector_type(16)));
typedef _Float16 v8h  __attribute__((ext_vector_type(8)));
typedef _Float16 v4h  __attribute__((ext_vector_type(4)));
typedef float    v8f  __attribute__((ext_vector_type(8)));
typedef float    v4f  __attribute__((ext_vector_type(4)));

// LDS strides (in elements), padded for bank-conflict-free b128 access
#define XS 200   // xh / O rows (192 + 8)
#define QS 408   // q||k rows (384 + 24)
#define VS 72    // vT rows (64 tokens + 8)
#define MS 68    // msum rows (f32)
#define SS 68    // S rows (f32)
#define PS 72    // P rows (f16)

struct __align__(16) LDSLayout {
  _Float16 xh[TOK * XS];      // staged x (f16); reused as attention output O
  _Float16 qkh[TOK * QS];     // q at cols 0..191, k at cols 192..383
  _Float16 vT[NH * HD * VS];  // v transposed: [head][dim][token]
  float    msum[TOK * MS];    // sp_mask[w] + mask[w]
  float    S[TOK * SS];       // logits, one head at a time
  _Float16 P[TOK * PS];       // softmax probs (f16)
};

__device__ __forceinline__ v16h load_frag(const _Float16* rowk, int kb) {
  // 16-bit WMMA operand: lanes 0-15 hold K {0..7,16..23}, lanes 16-31 K {8..15,24..31}
  union { v16h v; v8h h[2]; } u;
  u.h[0] = *(const v8h*)(rowk + kb);
  u.h[1] = *(const v8h*)(rowk + kb + 16);
  return u.v;
}

__device__ __forceinline__ v8f zero8() {
  v8f z;
#pragma unroll
  for (int i = 0; i < 8; ++i) z[i] = 0.0f;
  return z;
}

#define WMMA_F16(a, b, c) \
  __builtin_amdgcn_wmma_f32_16x16x32_f16(false, (a), false, (b), (short)0, (c), false, false)

// ---------------- prep: f32->f16 weights + rpb gather ----------------
__global__ __launch_bounds__(256) void prep_kernel(
    const float* __restrict__ qkv_w, const float* __restrict__ proj_w,
    const float* __restrict__ rpb_table, const int* __restrict__ rpi,
    _Float16* __restrict__ qkvw_h, _Float16* __restrict__ projw_h,
    float* __restrict__ rpb) {
  const int T1 = 576 * 192, T2 = 192 * 192, T3 = NH * 64 * 64;
  int i = blockIdx.x * 256 + threadIdx.x;
  if (i < T1) {
    qkvw_h[i] = (_Float16)qkv_w[i];
  } else if (i < T1 + T2) {
    int j = i - T1;
    projw_h[j] = (_Float16)proj_w[j];
  } else if (i < T1 + T2 + T3) {
    int e = i - T1 - T2;
    int h = e / 4096, ij = e % 4096;
    rpb[e] = rpb_table[rpi[ij] * NH + h];
  }
}

// ---------------- fused attention, one window per block ----------------
__global__ __launch_bounds__(256, 1) void fused_win_attn(
    const float* __restrict__ x,
    const float* __restrict__ mask,
    const float* __restrict__ sp_mask,
    const float* __restrict__ qkv_b,
    const float* __restrict__ logit_scale,
    const float* __restrict__ proj_b,
    const _Float16* __restrict__ qkvw_h,
    const _Float16* __restrict__ projw_h,
    const float* __restrict__ rpb,
    float* __restrict__ out) {
  __shared__ LDSLayout L;
  const int tid    = threadIdx.x;
  const int wave   = tid >> 5;
  const int lane   = tid & 31;
  const int laneM  = lane & 15;
  const int laneHi = lane >> 4;
  const int kb     = laneHi * 8;
  const int b      = blockIdx.x;
  const int w      = b & 4095;   // both nP and nW are 4096

  // ---- stage x (f32 -> f16) and msum = sp_mask[w] + mask[w] ----
  {
    const float* xb = x + (size_t)b * (TOK * DIMC);
#pragma unroll
    for (int j = 0; j < 12; ++j) {
      int e = j * 1024 + tid * 4;            // 12288 f32 total
      v4f xv = *(const v4f*)(xb + e);
      v4h hv;
#pragma unroll
      for (int i = 0; i < 4; ++i) hv[i] = (_Float16)xv[i];
      *(v4h*)&L.xh[(e / 192) * XS + (e % 192)] = hv;
    }
    const float* mk = mask + (size_t)w * 4096;
    const float* sk = sp_mask + (size_t)w * 4096;
#pragma unroll
    for (int j = 0; j < 4; ++j) {
      int e = j * 1024 + tid * 4;            // 4096 f32 total
      v4f a  = *(const v4f*)(mk + e);
      v4f s2 = *(const v4f*)(sk + e);
      v4f r;
#pragma unroll
      for (int i = 0; i < 4; ++i) r[i] = a[i] + s2[i];
      *(v4f*)&L.msum[(e >> 6) * MS + (e & 63)] = r;
    }
  }
  __syncthreads();

  // ---- QKV GEMM: (64x192) @ (192x576), tiles 4(M) x 36(N), K-steps of 32 ----
  for (int t = wave; t < 144; t += 8) {
    int mt = t / 36, nt = t % 36;
    const _Float16* arow = &L.xh[(mt * 16 + laneM) * XS];
    const _Float16* brow = qkvw_h + (nt * 16 + laneM) * DIMC;
    v8f c = zero8();
#pragma unroll
    for (int kk = 0; kk < 6; ++kk) {
      v16h a  = load_frag(arow + kk * 32, kb);
      v16h bb = load_frag(brow + kk * 32, kb);
      c = WMMA_F16(a, bb, c);
    }
    int nglob = nt * 16 + laneM;
    float bv = qkv_b[nglob];
    if (nt < 24) {
      // q (0..191) or k (192..383) -> qkh[token][nglob]
#pragma unroll
      for (int r = 0; r < 8; ++r)
        L.qkh[(mt * 16 + r + 8 * laneHi) * QS + nglob] = (_Float16)(c[r] + bv);
    } else {
      // v -> transposed per head: vT[h][d][token]; 8 accum rows = contiguous tokens
      int hh = (nglob - 384) >> 5;
      int dl = (nglob - 384) & 31;
      v8h ph;
#pragma unroll
      for (int r = 0; r < 8; ++r) ph[r] = (_Float16)(c[r] + bv);
      *(v8h*)&L.vT[hh * (HD * VS) + dl * VS + mt * 16 + 8 * laneHi] = ph;
    }
  }
  __syncthreads();

  // ---- L2-normalize q,k rows; fold exp(min(logit_scale, ln100)) into q ----
  for (int j = tid; j < 768; j += 256) {   // 64 tok * 6 heads * {q,k}
    int tok  = j & 63;
    int rest = j >> 6;        // 0..11
    int h    = rest % 6;
    int isK  = rest / 6;
    _Float16* p = &L.qkh[tok * QS + isK * 192 + h * HD];
    union { v8h v; _Float16 e[8]; } u[4];
    float s = 0.0f;
#pragma unroll
    for (int q4 = 0; q4 < 4; ++q4) {
      u[q4].v = *(const v8h*)(p + 8 * q4);
#pragma unroll
      for (int i = 0; i < 8; ++i) { float v = (float)u[q4].e[i]; s += v * v; }
    }
    float f = 1.0f / fmaxf(sqrtf(s), 1e-12f);
    if (!isK) f *= __expf(fminf(logit_scale[h], 4.6051702f)); // ln(100)
#pragma unroll
    for (int q4 = 0; q4 < 4; ++q4) {
#pragma unroll
      for (int i = 0; i < 8; ++i) u[q4].e[i] = (_Float16)((float)u[q4].e[i] * f);
      *(v8h*)(p + 8 * q4) = u[q4].v;
    }
  }
  __syncthreads();

  // ---- per-head attention ----
  for (int h = 0; h < NH; ++h) {
    // S = (qn*scale) @ kn^T : 16 tiles of 16x16, single K=32 step
    for (int t = wave; t < 16; t += 8) {
      int mt = t >> 2, nt = t & 3;
      v16h a  = load_frag(&L.qkh[(mt * 16 + laneM) * QS + h * HD], kb);
      v16h bb = load_frag(&L.qkh[(nt * 16 + laneM) * QS + 192 + h * HD], kb);
      v8f c = zero8();
      c = WMMA_F16(a, bb, c);
#pragma unroll
      for (int r = 0; r < 8; ++r)
        L.S[(mt * 16 + r + 8 * laneHi) * SS + nt * 16 + laneM] = c[r];
    }
    __syncthreads();

    // softmax over rows: 4 adjacent lanes per row, +rpb +msum fused in
    {
      int row = tid >> 2;
      int part = tid & 3;
      const float* srow = &L.S[row * SS + part * 16];
      const float* mrow = &L.msum[row * MS + part * 16];
      const float* rrow = rpb + h * 4096 + row * 64 + part * 16;
      float vals[16];
#pragma unroll
      for (int q4 = 0; q4 < 4; ++q4) {
        v4f sv = *(const v4f*)(srow + 4 * q4);
        v4f mv = *(const v4f*)(mrow + 4 * q4);
        v4f rv = *(const v4f*)(rrow + 4 * q4);
#pragma unroll
        for (int i = 0; i < 4; ++i) vals[q4 * 4 + i] = sv[i] + mv[i] + rv[i];
      }
      float m = vals[0];
#pragma unroll
      for (int i = 1; i < 16; ++i) m = fmaxf(m, vals[i]);
      m = fmaxf(m, __shfl_xor(m, 1, 32));
      m = fmaxf(m, __shfl_xor(m, 2, 32));
      float s = 0.0f;
#pragma unroll
      for (int i = 0; i < 16; ++i) { vals[i] = __expf(vals[i] - m); s += vals[i]; }
      s += __shfl_xor(s, 1, 32);
      s += __shfl_xor(s, 2, 32);
      float inv = 1.0f / s;
      v8h p0, p1;
#pragma unroll
      for (int i = 0; i < 8; ++i) {
        p0[i] = (_Float16)(vals[i] * inv);
        p1[i] = (_Float16)(vals[8 + i] * inv);
      }
      *(v8h*)&L.P[row * PS + part * 16]     = p0;
      *(v8h*)&L.P[row * PS + part * 16 + 8] = p1;
    }
    __syncthreads();

    // O_h = P @ V_h : 8 tiles (4 M x 2 D), K=64 in two steps; O lives in L.xh
    {
      int mt = wave >> 1, dt = wave & 1;
      v8f c = zero8();
#pragma unroll
      for (int kk = 0; kk < 2; ++kk) {
        v16h a  = load_frag(&L.P[(mt * 16 + laneM) * PS + kk * 32], kb);
        v16h bb = load_frag(&L.vT[h * (HD * VS) + (dt * 16 + laneM) * VS + kk * 32], kb);
        c = WMMA_F16(a, bb, c);
      }
#pragma unroll
      for (int r = 0; r < 8; ++r)
        L.xh[(mt * 16 + r + 8 * laneHi) * XS + h * HD + dt * 16 + laneM] = (_Float16)c[r];
    }
    __syncthreads();
  }

  // ---- proj GEMM: (64x192) @ (192x192) + bias -> out ----
  float* ob = out + (size_t)b * (TOK * DIMC);
  for (int t = wave; t < 48; t += 8) {
    int mt = t / 12, nt = t % 12;
    const _Float16* arow = &L.xh[(mt * 16 + laneM) * XS];
    const _Float16* brow = projw_h + (nt * 16 + laneM) * DIMC;
    v8f c = zero8();
#pragma unroll
    for (int kk = 0; kk < 6; ++kk) {
      v16h a  = load_frag(arow + kk * 32, kb);
      v16h bb = load_frag(brow + kk * 32, kb);
      c = WMMA_F16(a, bb, c);
    }
    float pb = proj_b[nt * 16 + laneM];
#pragma unroll
    for (int r = 0; r < 8; ++r)
      ob[(mt * 16 + r + 8 * laneHi) * DIMC + nt * 16 + laneM] = c[r] + pb;
  }
}

extern "C" void kernel_launch(void* const* d_in, const int* in_sizes, int n_in,
                              void* d_out, int out_size, void* d_ws, size_t ws_size,
                              hipStream_t stream) {
  const float* x           = (const float*)d_in[0];
  const int*   rpi         = (const int*)d_in[1];
  const float* mask        = (const float*)d_in[2];
  const float* sp_mask     = (const float*)d_in[3];
  const float* qkv_w       = (const float*)d_in[4];
  const float* qkv_b       = (const float*)d_in[5];
  const float* logit_scale = (const float*)d_in[6];
  const float* rpb_table   = (const float*)d_in[7];
  const float* proj_w      = (const float*)d_in[8];
  const float* proj_b      = (const float*)d_in[9];
  float* out = (float*)d_out;

  // workspace layout: f16 qkv_w (221184 B) | f16 proj_w (73728 B) | rpb f32 (98304 B)
  _Float16* qkvw_h  = (_Float16*)d_ws;
  _Float16* projw_h = (_Float16*)((char*)d_ws + 221184);
  float*    rpb     = (float*)((char*)d_ws + 294912);

  prep_kernel<<<672, 256, 0, stream>>>(qkv_w, proj_w, rpb_table, rpi, qkvw_h, projw_h, rpb);
  fused_win_attn<<<8192, 256, 0, stream>>>(x, mask, sp_mask, qkv_b, logit_scale,
                                           proj_b, qkvw_h, projw_h, rpb, out);
}
